// InGramRelationLayer_52003464019983
// MI455X (gfx1250) — matrix operand order repaired and checked by reading, not compiled, not run on hardware
//
#include <hip/hip_runtime.h>
#include <math.h>

// Problem constants (also derived from in_sizes at launch)
#define DIM_IN   128
#define DIM_OUT  128
#define NUM_HEAD 8
#define WAVES    8     // waves per block (256 threads, wave32)

typedef __bf16 v16bf __attribute__((ext_vector_type(16)));
typedef __bf16 v8bf  __attribute__((ext_vector_type(8)));
typedef float  v8f   __attribute__((ext_vector_type(8)));

union BF16x16 { v16bf v; v8bf h[2]; };

__device__ __forceinline__ __bf16 f2bf(float f) {
  unsigned u = __float_as_uint(f);
  unsigned r = (u + 0x7FFFu + ((u >> 16) & 1u)) >> 16;   // round to nearest even
  unsigned short s = (unsigned short)r;
  return __builtin_bit_cast(__bf16, s);
}
__device__ __forceinline__ float bf2f(__bf16 b) {
  unsigned short s = __builtin_bit_cast(unsigned short, b);
  return __uint_as_float(((unsigned)s) << 16);
}

// A fragment (16x32 bf16, MxK): lane L holds row M=L&15.
// Lanes 0-15: K = kbase+0..15 ; lanes 16-31: K = kbase+16..31 (contiguous).
__device__ __forceinline__ v16bf load_a_frag(const __bf16* __restrict__ row, int kbase, int sel) {
  BF16x16 u;
  u.h[0] = *reinterpret_cast<const v8bf*>(row + kbase + sel * 16);
  u.h[1] = *reinterpret_cast<const v8bf*>(row + kbase + sel * 16 + 8);
  return u.v;
}

// B fragment (32x16 bf16, KxN): lane L holds column N=L&15, K = (L>>4)*16 + 0..15 contiguous.
// B[k][n] = W[ncol][kdim0+k] -> 16 contiguous bf16 from the weight row.
__device__ __forceinline__ v16bf load_b_frag(const __bf16* __restrict__ wrow, int kdim0) {
  return *reinterpret_cast<const v16bf*>(wrow + kdim0);
}

__device__ __forceinline__ void atomic_max_f32(float* addr, float val) {
  if (val >= 0.0f) atomicMax((int*)addr, __float_as_int(val));
  else             atomicMin((unsigned int*)addr, __float_as_uint(val));
}

// -------------------- init / convert --------------------
__global__ void k_init(float* __restrict__ out, float* __restrict__ m_ws,
                       float* __restrict__ s_ws, int out_n, int ms_n) {
  int i = blockIdx.x * blockDim.x + threadIdx.x;
  if (i < out_n) out[i] = 0.0f;
  if (i < ms_n) { m_ws[i] = -3.0e38f; s_ws[i] = 0.0f; }
}

__global__ void k_split_bf16(const float* __restrict__ src, __bf16* __restrict__ hi,
                             __bf16* __restrict__ lo, int n) {
  int i = blockIdx.x * blockDim.x + threadIdx.x;
  if (i >= n) return;
  float f = src[i];
  __bf16 h = f2bf(f);
  hi[i] = h;
  lo[i] = f2bf(f - bf2f(h));
}

// -------------------- pass A: attention logits + segment max --------------------
__global__ void __launch_bounds__(256) k_attn_raw(
    const __bf16* __restrict__ emb_hi, const __bf16* __restrict__ emb_lo,
    const __bf16* __restrict__ aw_hi,  const __bf16* __restrict__ aw_lo,
    const float* __restrict__ attn_b,  const float* __restrict__ attn_vec,
    const int* __restrict__ trip,
    float* __restrict__ raw_ws, float* __restrict__ m_ws, int num_tiles) {
  __shared__ float raw_lds[WAVES][16][NUM_HEAD];
  const int wave = threadIdx.x >> 5;
  const int lane = threadIdx.x & 31;
  const int tile = blockIdx.x * WAVES + wave;
  if (tile >= num_tiles) return;
  const int e0   = tile * 16;
  const int erow = lane & 15;
  const int sel  = lane >> 4;
  const int eidx = e0 + erow;
  const int hrow = trip[eidx * 3 + 0];
  const int trow = trip[eidx * 3 + 1];

  v8f acc[NUM_HEAD] = {};

  #pragma unroll
  for (int src = 0; src < 2; ++src) {
    const int rowid = src ? trow : hrow;
    const __bf16* rh = emb_hi + (size_t)rowid * DIM_IN;
    const __bf16* rl = emb_lo + (size_t)rowid * DIM_IN;
    #pragma unroll
    for (int kb = 0; kb < 4; ++kb) {
      v16bf Ah = load_a_frag(rh, kb * 32, sel);
      v16bf Al = load_a_frag(rl, kb * 32, sel);
      const int kdim = src * DIM_IN + kb * 32 + sel * 16;
      #pragma unroll
      for (int nt = 0; nt < NUM_HEAD; ++nt) {
        const __bf16* wh = aw_hi + (size_t)(nt * 16 + erow) * (2 * DIM_IN);
        const __bf16* wl = aw_lo + (size_t)(nt * 16 + erow) * (2 * DIM_IN);
        v16bf Bh = load_b_frag(wh, kdim);
        v16bf Bl = load_b_frag(wl, kdim);
        acc[nt] = __builtin_amdgcn_wmma_f32_16x16x32_bf16(false, Ah, false, Bh, (short)0, acc[nt], false, false);
        acc[nt] = __builtin_amdgcn_wmma_f32_16x16x32_bf16(false, Ah, false, Bl, (short)0, acc[nt], false, false);
        acc[nt] = __builtin_amdgcn_wmma_f32_16x16x32_bf16(false, Al, false, Bh, (short)0, acc[nt], false, false);
      }
    }
  }

  // tanh, * attn_vec, reduce over N (16 lanes per half), stage to LDS
  #pragma unroll
  for (int nt = 0; nt < NUM_HEAD; ++nt) {
    const float bc = attn_b[nt * 16 + erow];
    const float vc = attn_vec[nt * 16 + erow];
    float v[8];
    #pragma unroll
    for (int r = 0; r < 8; ++r) v[r] = tanhf(acc[nt][r] + bc) * vc;
    #pragma unroll
    for (int m = 1; m < 16; m <<= 1) {
      #pragma unroll
      for (int r = 0; r < 8; ++r) v[r] += __shfl_xor(v[r], m, 32);
    }
    if (erow == 0) {
      #pragma unroll
      for (int r = 0; r < 8; ++r) raw_lds[wave][sel * 8 + r][nt] = v[r];
    }
  }
  asm volatile("s_wait_dscnt 0x0" ::: "memory");
  if (lane < 16) {
    #pragma unroll
    for (int hd = 0; hd < NUM_HEAD; ++hd) {
      float rv = raw_lds[wave][lane][hd];
      raw_ws[(size_t)eidx * NUM_HEAD + hd] = rv;
      atomic_max_f32(&m_ws[hrow * NUM_HEAD + hd], rv);
    }
  }
}

// -------------------- pass B: exp / segment sum / beta --------------------
__global__ void k_softmax_ev(const int* __restrict__ trip, const float* __restrict__ m_ws,
                             float* __restrict__ raw_ws, float* __restrict__ s_ws, int n) {
  int i = blockIdx.x * blockDim.x + threadIdx.x;
  if (i >= n) return;
  int e = i >> 3, hd = i & 7;
  int h = trip[e * 3];
  float ev = expf(raw_ws[i] - m_ws[h * NUM_HEAD + hd]);
  raw_ws[i] = ev;
  atomicAdd(&s_ws[h * NUM_HEAD + hd], ev);
}

__global__ void k_softmax_beta(const int* __restrict__ trip, const float* __restrict__ s_ws,
                               float* __restrict__ raw_ws, int n) {
  int i = blockIdx.x * blockDim.x + threadIdx.x;
  if (i >= n) return;
  int e = i >> 3, hd = i & 7;
  int h = trip[e * 3];
  raw_ws[i] = raw_ws[i] / (s_ws[h * NUM_HEAD + hd] + 1e-16f);
}

// -------------------- pass C: messages + weighted segment sum --------------------
__global__ void __launch_bounds__(256) k_aggregate(
    const __bf16* __restrict__ emb_hi, const __bf16* __restrict__ emb_lo,
    const __bf16* __restrict__ gw_hi,  const __bf16* __restrict__ gw_lo,
    const float* __restrict__ aggr_b,
    const int* __restrict__ trip, const float* __restrict__ beta_ws,
    float* __restrict__ out, int num_tiles) {
  __shared__ int head_lds[WAVES][16];
  const int wave = threadIdx.x >> 5;
  const int lane = threadIdx.x & 31;
  const int tile = blockIdx.x * WAVES + wave;
  if (tile >= num_tiles) return;
  const int e0   = tile * 16;
  const int erow = lane & 15;
  const int sel  = lane >> 4;
  const int eidx = e0 + erow;
  const int hrow = trip[eidx * 3 + 0];
  const int trow = trip[eidx * 3 + 1];
  if (lane < 16) head_lds[wave][lane] = hrow;
  asm volatile("s_wait_dscnt 0x0" ::: "memory");

  v8f acc[NUM_HEAD] = {};
  const __bf16* rh = emb_hi + (size_t)trow * DIM_IN;
  const __bf16* rl = emb_lo + (size_t)trow * DIM_IN;
  #pragma unroll
  for (int kb = 0; kb < 4; ++kb) {
    v16bf Ah = load_a_frag(rh, kb * 32, sel);
    v16bf Al = load_a_frag(rl, kb * 32, sel);
    const int kdim = kb * 32 + sel * 16;
    #pragma unroll
    for (int nt = 0; nt < NUM_HEAD; ++nt) {
      const __bf16* wh = gw_hi + (size_t)(nt * 16 + erow) * DIM_IN;
      const __bf16* wl = gw_lo + (size_t)(nt * 16 + erow) * DIM_IN;
      v16bf Bh = load_b_frag(wh, kdim);
      v16bf Bl = load_b_frag(wl, kdim);
      acc[nt] = __builtin_amdgcn_wmma_f32_16x16x32_bf16(false, Ah, false, Bh, (short)0, acc[nt], false, false);
      acc[nt] = __builtin_amdgcn_wmma_f32_16x16x32_bf16(false, Ah, false, Bl, (short)0, acc[nt], false, false);
      acc[nt] = __builtin_amdgcn_wmma_f32_16x16x32_bf16(false, Al, false, Bh, (short)0, acc[nt], false, false);
    }
  }

  #pragma unroll
  for (int nt = 0; nt < NUM_HEAD; ++nt) {
    const float bc = aggr_b[nt * 16 + erow];
    #pragma unroll
    for (int r = 0; r < 8; ++r) {
      int M = sel * 8 + r;
      float beta = beta_ws[(size_t)(e0 + M) * NUM_HEAD + nt];
      int h = head_lds[wave][M];
      atomicAdd(&out[(size_t)h * DIM_OUT + nt * 16 + erow], (acc[nt][r] + bc) * beta);
    }
  }
}

// -------------------- launch --------------------
extern "C" void kernel_launch(void* const* d_in, const int* in_sizes, int n_in,
                              void* d_out, int out_size, void* d_ws, size_t ws_size,
                              hipStream_t stream) {
  const float* emb_rel  = (const float*)d_in[0];
  const int*   trip     = (const int*)d_in[1];
  const float* attn_w   = (const float*)d_in[2];
  const float* attn_b   = (const float*)d_in[3];
  const float* attn_vec = (const float*)d_in[4];
  const float* aggr_w   = (const float*)d_in[5];
  const float* aggr_b   = (const float*)d_in[6];
  float* out = (float*)d_out;

  const int n_emb     = in_sizes[0];            // 50000*128
  const int num_rel   = n_emb / DIM_IN;
  const int num_edges = in_sizes[1] / 3;        // 1,000,000
  const int n_aw      = in_sizes[2];            // 128*256
  const int n_gw      = in_sizes[5];            // 128*128

  char* ws = (char*)d_ws;
  size_t off = 0;
  auto take = [&](size_t bytes) -> char* {
    char* p = ws + off;
    off = (off + bytes + 255) & ~(size_t)255;
    return p;
  };
  __bf16* emb_hi = (__bf16*)take((size_t)n_emb * 2);
  __bf16* emb_lo = (__bf16*)take((size_t)n_emb * 2);
  __bf16* aw_hi  = (__bf16*)take((size_t)n_aw * 2);
  __bf16* aw_lo  = (__bf16*)take((size_t)n_aw * 2);
  __bf16* gw_hi  = (__bf16*)take((size_t)n_gw * 2);
  __bf16* gw_lo  = (__bf16*)take((size_t)n_gw * 2);
  float*  raw_ws = (float*)take((size_t)num_edges * NUM_HEAD * 4);  // raw -> ev -> beta
  float*  m_ws   = (float*)take((size_t)num_rel * NUM_HEAD * 4);
  float*  s_ws   = (float*)take((size_t)num_rel * NUM_HEAD * 4);
  (void)ws_size; (void)n_in;

  const int ms_n   = num_rel * NUM_HEAD;
  const int init_n = out_size > ms_n ? out_size : ms_n;
  k_init<<<(init_n + 255) / 256, 256, 0, stream>>>(out, m_ws, s_ws, out_size, ms_n);

  k_split_bf16<<<(n_emb + 255) / 256, 256, 0, stream>>>(emb_rel, emb_hi, emb_lo, n_emb);
  k_split_bf16<<<(n_aw  + 255) / 256, 256, 0, stream>>>(attn_w,  aw_hi,  aw_lo,  n_aw);
  k_split_bf16<<<(n_gw  + 255) / 256, 256, 0, stream>>>(aggr_w,  gw_hi,  gw_lo,  n_gw);

  const int tiles  = num_edges / 16;            // 62500 (exact)
  const int blocks = (tiles + WAVES - 1) / WAVES;
  k_attn_raw<<<blocks, 256, 0, stream>>>(emb_hi, emb_lo, aw_hi, aw_lo,
                                         attn_b, attn_vec, trip, raw_ws, m_ws, tiles);

  const int n8 = num_edges * NUM_HEAD;
  k_softmax_ev<<<(n8 + 255) / 256, 256, 0, stream>>>(trip, m_ws, raw_ws, s_ws, n8);
  k_softmax_beta<<<(n8 + 255) / 256, 256, 0, stream>>>(trip, s_ws, raw_ws, n8);

  k_aggregate<<<blocks, 256, 0, stream>>>(emb_hi, emb_lo, gw_hi, gw_lo, aggr_b,
                                          trip, raw_ws, out, tiles);
}
